// AttentionLayer_67997922230859
// MI455X (gfx1250) — matrix operand order repaired and checked by reading, not compiled, not run on hardware
//
#include <hip/hip_runtime.h>

// ---------------------------------------------------------------------------
// Transformer encoder layer forward for MI455X (gfx1250, wave32, WMMA).
// All GEMMs use v_wmma_f32_16x16x32_bf16 (bf16 inputs, fp32 accumulate).
// Software-pipelined: next K-step fragments are loaded before current WMMAs.
// ---------------------------------------------------------------------------

typedef __attribute__((ext_vector_type(16))) __bf16 v16bf;
typedef __attribute__((ext_vector_type(8)))  __bf16 v8bf;
typedef __attribute__((ext_vector_type(8)))  float  v8f;

#define D_MODEL 1024
#define SEQ_L   2048
#define N_HEAD  16
#define D_HEAD  64
#define FFN_DIM 4096
#define BATCH   2
#define M_ROWS  (BATCH * SEQ_L)   // 4096

union FragU { v16bf v; v8bf h[2]; };

__device__ __forceinline__ v16bf ld_frag2(const __bf16* p0, const __bf16* p1) {
    FragU f;
    f.h[0] = *(const v8bf*)p0;
    f.h[1] = *(const v8bf*)p1;
    return f.v;
}

__device__ __forceinline__ v8f vzero() {
    v8f z;
#pragma unroll
    for (int i = 0; i < 8; ++i) z[i] = 0.0f;
    return z;
}

// Compiler-only ordering fence: LDS producer/consumer is intra-wave (DS ops
// are in-order per wave on CDNA5), we only need to stop compiler reordering.
__device__ __forceinline__ void lds_fence() { asm volatile("" ::: "memory"); }

__device__ __forceinline__ float gelu_exact(float x) {
    return 0.5f * x * (1.0f + erff(x * 0.70710678118654752f));
}

#define WMMA_BF16(A, B, C) \
    __builtin_amdgcn_wmma_f32_16x16x32_bf16(false, (A), false, (B), (short)0, (C), false, false)

// ---------------------------------------------------------------------------
// Elementwise fp32 -> bf16 convert
// ---------------------------------------------------------------------------
__global__ void cvt_bf16_kernel(const float* __restrict__ in, __bf16* __restrict__ out, int n) {
    int i = blockIdx.x * blockDim.x + threadIdx.x;
    if (i < n) out[i] = (__bf16)in[i];
}

// ---------------------------------------------------------------------------
// Weight transpose + convert: w[K][N] fp32 -> wt[N][K] bf16  (tiled, coalesced)
// ---------------------------------------------------------------------------
__global__ void __launch_bounds__(256) wtrans_kernel(const float* __restrict__ w,
                                                     __bf16* __restrict__ wt,
                                                     int K, int N) {
    __shared__ float tile[32][33];
    int nb = blockIdx.x * 32, kb = blockIdx.y * 32;
    int tx = threadIdx.x, ty = threadIdx.y;
#pragma unroll
    for (int i = 0; i < 4; ++i)
        tile[ty + i * 8][tx] = w[(size_t)(kb + ty + i * 8) * N + nb + tx];
    __syncthreads();
#pragma unroll
    for (int i = 0; i < 4; ++i)
        wt[(size_t)(nb + ty + i * 8) * K + kb + tx] = (__bf16)tile[tx][ty + i * 8];
}

// ---------------------------------------------------------------------------
// Per-head V transpose: v[B*L][D] bf16 -> vt[(b*H+h)*64 + d][L] bf16
// ---------------------------------------------------------------------------
__global__ void __launch_bounds__(256) vtrans_kernel(const __bf16* __restrict__ v,
                                                     __bf16* __restrict__ vt) {
    __shared__ __bf16 tile[32][34];
    int lb = blockIdx.x * 32, db = blockIdx.y * 32, bh = blockIdx.z;
    int b = bh >> 4, h = bh & 15;
    int tx = threadIdx.x, ty = threadIdx.y;
#pragma unroll
    for (int i = 0; i < 4; ++i)
        tile[ty + i * 8][tx] =
            v[(size_t)(b * SEQ_L + lb + ty + i * 8) * D_MODEL + h * D_HEAD + db + tx];
    __syncthreads();
#pragma unroll
    for (int i = 0; i < 4; ++i)
        vt[((size_t)bh * D_HEAD + db + ty + i * 8) * SEQ_L + lb + tx] = tile[tx][ty + i * 8];
}

// ---------------------------------------------------------------------------
// WMMA GEMM: C[M][N] = A[M][K](bf16) @ Bt[N][K]^T(bf16) + bias, optional GELU.
// Block 256 thr = 8 waves (4x2), wave tile 32x64, block tile 128x128.
// Double-buffered fragments: loads for k+32 are issued before WMMAs for k.
// grid(N/128, M/128).  EPI: 0=none 1=gelu.  OUTB: 1=bf16 out, 0=f32 out.
// ---------------------------------------------------------------------------
template <int EPI, int OUTB>
__global__ void __launch_bounds__(256) gemm_kernel(const __bf16* __restrict__ A,
                                                   const __bf16* __restrict__ Bt,
                                                   const float* __restrict__ bias,
                                                   float* __restrict__ Cf,
                                                   __bf16* __restrict__ Cb,
                                                   int M, int N, int K) {
    const int lane = threadIdx.x & 31;
    const int w    = threadIdx.x >> 5;
    const int wm   = w >> 1, wn = w & 1;
    const int m0   = blockIdx.y * 128 + wm * 32;   // 2 M-tiles
    const int n0   = blockIdx.x * 128 + wn * 64;   // 4 N-tiles
    const int rlo  = lane & 15;
    const int half = lane >> 4;

    v8f acc[2][4];
#pragma unroll
    for (int i = 0; i < 2; ++i)
#pragma unroll
        for (int j = 0; j < 4; ++j) acc[i][j] = vzero();

    const __bf16* arow[2];
    const __bf16* brow[4];
    arow[0] = A + (size_t)(m0 + rlo) * K;
    arow[1] = A + (size_t)(m0 + 16 + rlo) * K;
#pragma unroll
    for (int j = 0; j < 4; ++j) brow[j] = Bt + (size_t)(n0 + j * 16 + rlo) * K;

    const int ka = half * 8;   // A frag: elems 0..7 -> K=ka.., elems 8..15 -> K=ka+16..
    const int kb = half * 8;   // B frag: elems 0..7 -> K=kb.., elems 8..15 -> K=kb+16..

    v16bf af[2], bf[4];
#pragma unroll
    for (int i = 0; i < 2; ++i)
        af[i] = ld_frag2(arow[i] + ka, arow[i] + ka + 16);
#pragma unroll
    for (int j = 0; j < 4; ++j)
        bf[j] = ld_frag2(brow[j] + kb, brow[j] + kb + 16);

    for (int k = 0; k + 32 < K; k += 32) {
        // issue next K-step loads first (fresh registers -> overlap with WMMAs)
        v16bf afn[2], bfn[4];
        const int kn = k + 32;
#pragma unroll
        for (int i = 0; i < 2; ++i)
            afn[i] = ld_frag2(arow[i] + kn + ka, arow[i] + kn + ka + 16);
#pragma unroll
        for (int j = 0; j < 4; ++j)
            bfn[j] = ld_frag2(brow[j] + kn + kb, brow[j] + kn + kb + 16);

#pragma unroll
        for (int i = 0; i < 2; ++i)
#pragma unroll
            for (int j = 0; j < 4; ++j)
                acc[i][j] = WMMA_BF16(af[i], bf[j], acc[i][j]);

#pragma unroll
        for (int i = 0; i < 2; ++i) af[i] = afn[i];
#pragma unroll
        for (int j = 0; j < 4; ++j) bf[j] = bfn[j];
    }
    // tail step
#pragma unroll
    for (int i = 0; i < 2; ++i)
#pragma unroll
        for (int j = 0; j < 4; ++j)
            acc[i][j] = WMMA_BF16(af[i], bf[j], acc[i][j]);

#pragma unroll
    for (int i = 0; i < 2; ++i) {
#pragma unroll
        for (int j = 0; j < 4; ++j) {
            const int col = n0 + j * 16 + rlo;
            const float bv = bias[col];
#pragma unroll
            for (int r = 0; r < 8; ++r) {
                const int row = m0 + i * 16 + half * 8 + r;
                float val = acc[i][j][r] + bv;
                if (EPI == 1) val = gelu_exact(val);
                if (OUTB) Cb[(size_t)row * N + col] = (__bf16)val;
                else      Cf[(size_t)row * N + col] = val;
            }
        }
    }
}

// ---------------------------------------------------------------------------
// Flash attention (no mask): one wave per 16-query tile, 32-key chunks.
// q,k: [B*L][D] bf16 (head slice at h*64); vt: [(b*H+h)*64 + d][L] bf16.
// o: [B*L][D] bf16.  grid(B*H, L/64), block 128 (4 waves).
// Next chunk's K/V fragments prefetched so softmax VALU overlaps global loads.
// LDS regions are strictly per-wave -> no block barriers needed.
// ---------------------------------------------------------------------------
__global__ void __launch_bounds__(128) attn_kernel(const __bf16* __restrict__ q,
                                                   const __bf16* __restrict__ kk,
                                                   const __bf16* __restrict__ vt,
                                                   __bf16* __restrict__ o) {
    __shared__ float  S[4][16][32];
    __shared__ __bf16 P[4][16][32];
    __shared__ float  CB[4][16];
    __shared__ float  LI[4][16];

    const int w    = threadIdx.x >> 5;
    const int lane = threadIdx.x & 31;
    const int bh   = blockIdx.x;
    const int b    = bh >> 4, h = bh & 15;
    const int q0   = blockIdx.y * 64 + w * 16;
    const int rlo  = lane & 15;
    const int half = lane >> 4;
    const float scale = 0.125f; // 1/sqrt(64)

    // Q fragments for d = 0..31 and 32..63 (loaded once)
    const __bf16* qrow = q + (size_t)(b * SEQ_L + q0 + rlo) * D_MODEL + h * D_HEAD;
    const v16bf qa0 = ld_frag2(qrow + half * 8,      qrow + half * 8 + 16);
    const v16bf qa1 = ld_frag2(qrow + 32 + half * 8, qrow + 32 + half * 8 + 16);

    v8f acc[4];
#pragma unroll
    for (int t = 0; t < 4; ++t) acc[t] = vzero();
    float m = -INFINITY, lsum = 0.0f;

    const __bf16* kbase = kk + (size_t)(b * SEQ_L) * D_MODEL + h * D_HEAD;
    const __bf16* vbase = vt + (size_t)bh * D_HEAD * SEQ_L;

    auto load_kf = [&](int kc, v16bf kf[2][2]) {
#pragma unroll
        for (int t = 0; t < 2; ++t) {
            const __bf16* krow = kbase + (size_t)(kc + t * 16 + rlo) * D_MODEL;
            kf[t][0] = ld_frag2(krow + half * 8,      krow + half * 8 + 16);
            kf[t][1] = ld_frag2(krow + 32 + half * 8, krow + 32 + half * 8 + 16);
        }
    };
    auto load_vf = [&](int kc, v16bf vf[4]) {
#pragma unroll
        for (int t = 0; t < 4; ++t) {
            const __bf16* vrow = vbase + (size_t)(t * 16 + rlo) * SEQ_L + kc + half * 8;
            vf[t] = ld_frag2(vrow, vrow + 16);
        }
    };

    v16bf kf[2][2], vf[4];
    load_kf(0, kf);
    load_vf(0, vf);

    for (int kc = 0; kc < SEQ_L; kc += 32) {
        // ---- S = scale * Q @ K^T  (two 16x16 key tiles, K-dim = 64 = 2x32)
#pragma unroll
        for (int t = 0; t < 2; ++t) {
            v8f st = vzero();
            st = WMMA_BF16(qa0, kf[t][0], st);
            st = WMMA_BF16(qa1, kf[t][1], st);
#pragma unroll
            for (int r = 0; r < 8; ++r)
                S[w][half * 8 + r][t * 16 + rlo] = st[r] * scale;
        }
        lds_fence();

        // ---- prefetch next chunk's K/V fragments (overlaps softmax VALU work)
        v16bf kf2[2][2], vf2[4];
        const bool more = (kc + 32) < SEQ_L;
        if (more) {
            load_kf(kc + 32, kf2);
            load_vf(kc + 32, vf2);
        }

        // ---- online softmax: lane pair (l, l+16) handles row l&15, 16 cols each
        {
            const int row = rlo, cb = half * 16;
            float mx = -INFINITY;
#pragma unroll
            for (int j = 0; j < 16; ++j) mx = fmaxf(mx, S[w][row][cb + j]);
            mx = fmaxf(mx, __shfl_xor(mx, 16));
            const float mnew = fmaxf(m, mx);
            const float corr = __expf(m - mnew);
            float ps = 0.0f;
#pragma unroll
            for (int j = 0; j < 16; ++j) {
                float p = __expf(S[w][row][cb + j] - mnew);
                ps += p;
                P[w][row][cb + j] = (__bf16)p;
            }
            ps += __shfl_xor(ps, 16);
            lsum = lsum * corr + ps;
            m = mnew;
            if (half == 0) CB[w][row] = corr;
        }
        lds_fence();

        // ---- rescale accumulators by per-row correction
#pragma unroll
        for (int r = 0; r < 8; ++r) {
            const float c = CB[w][half * 8 + r];
#pragma unroll
            for (int t = 0; t < 4; ++t) acc[t][r] *= c;
        }

        // ---- O += P @ V   (P: 16x32 A-frag from LDS; V fragments preloaded)
        const __bf16* prow = &P[w][rlo][0];
        v16bf pa = ld_frag2(prow + half * 8, prow + half * 8 + 16);
#pragma unroll
        for (int t = 0; t < 4; ++t)
            acc[t] = WMMA_BF16(pa, vf[t], acc[t]);

        if (more) {
#pragma unroll
            for (int t = 0; t < 2; ++t) { kf[t][0] = kf2[t][0]; kf[t][1] = kf2[t][1]; }
#pragma unroll
            for (int t = 0; t < 4; ++t) vf[t] = vf2[t];
        }
    }

    if (half == 0) LI[w][rlo] = 1.0f / lsum;
    lds_fence();

#pragma unroll
    for (int r = 0; r < 8; ++r) {
        const float sc = LI[w][half * 8 + r];
        const size_t orow = (size_t)(b * SEQ_L + q0 + half * 8 + r) * D_MODEL + h * D_HEAD;
#pragma unroll
        for (int t = 0; t < 4; ++t)
            o[orow + t * 16 + rlo] = (__bf16)(acc[t][r] * sc);
    }
}

// ---------------------------------------------------------------------------
// Residual + LayerNorm over D=1024. One block (256 thr) per row.
// ---------------------------------------------------------------------------
__global__ void __launch_bounds__(256) ln_kernel(const float* __restrict__ a,
                                                 const float* __restrict__ res,
                                                 const float* __restrict__ g,
                                                 const float* __restrict__ be,
                                                 float* __restrict__ outf,
                                                 __bf16* __restrict__ outb) {
    __shared__ float red[256];
    const int row = blockIdx.x, t = threadIdx.x;
    const float* pa = a   + (size_t)row * D_MODEL;
    const float* pr = res + (size_t)row * D_MODEL;

    float s[4];
    float partial = 0.0f;
#pragma unroll
    for (int i = 0; i < 4; ++i) {
        s[i] = pa[t + i * 256] + pr[t + i * 256];
        partial += s[i];
    }
    red[t] = partial;
    __syncthreads();
    for (int st = 128; st > 0; st >>= 1) {
        if (t < st) red[t] += red[t + st];
        __syncthreads();
    }
    const float mu = red[0] * (1.0f / D_MODEL);
    __syncthreads();

    float vp = 0.0f;
#pragma unroll
    for (int i = 0; i < 4; ++i) { float d = s[i] - mu; vp += d * d; }
    red[t] = vp;
    __syncthreads();
    for (int st = 128; st > 0; st >>= 1) {
        if (t < st) red[t] += red[t + st];
        __syncthreads();
    }
    const float rs = rsqrtf(red[0] * (1.0f / D_MODEL) + 1e-5f);

#pragma unroll
    for (int i = 0; i < 4; ++i) {
        const int c = t + i * 256;
        const float y = (s[i] - mu) * rs * g[c] + be[c];
        outf[(size_t)row * D_MODEL + c] = y;
        if (outb) outb[(size_t)row * D_MODEL + c] = (__bf16)y;
    }
}

// ---------------------------------------------------------------------------
// Host-side pipeline
// ---------------------------------------------------------------------------
extern "C" void kernel_launch(void* const* d_in, const int* in_sizes, int n_in,
                              void* d_out, int out_size, void* d_ws, size_t ws_size,
                              hipStream_t stream) {
    (void)in_sizes; (void)n_in; (void)out_size;

    const float* x   = (const float*)d_in[0];
    const float* wq  = (const float*)d_in[1];
    const float* bq  = (const float*)d_in[2];
    const float* wk  = (const float*)d_in[3];
    const float* bk  = (const float*)d_in[4];
    const float* wv  = (const float*)d_in[5];
    const float* bv  = (const float*)d_in[6];
    const float* wo  = (const float*)d_in[7];
    const float* bo  = (const float*)d_in[8];
    const float* g1  = (const float*)d_in[9];
    const float* b1  = (const float*)d_in[10];
    const float* g2  = (const float*)d_in[11];
    const float* b2  = (const float*)d_in[12];
    const float* w1  = (const float*)d_in[13];
    const float* bf1 = (const float*)d_in[14];
    const float* w2  = (const float*)d_in[15];
    const float* bf2 = (const float*)d_in[16];

    const size_t MB = 1ull << 20;
    if (ws_size < 128 * MB) return;

    char* ws = (char*)d_ws;
    __bf16* xb  = (__bf16*)(ws + 0);          // 8MB   (reused later as x1b)
    __bf16* wqT = (__bf16*)(ws + 8  * MB);    // 2MB
    __bf16* wkT = (__bf16*)(ws + 10 * MB);    // 2MB
    __bf16* wvT = (__bf16*)(ws + 12 * MB);    // 2MB
    __bf16* woT = (__bf16*)(ws + 14 * MB);    // 2MB
    __bf16* w1T = (__bf16*)(ws + 16 * MB);    // 8MB
    __bf16* w2T = (__bf16*)(ws + 24 * MB);    // 8MB
    __bf16* qb  = (__bf16*)(ws + 32 * MB);    // 8MB
    __bf16* kb  = (__bf16*)(ws + 40 * MB);    // 8MB
    __bf16* vb  = (__bf16*)(ws + 48 * MB);    // 8MB   (reused as ob)
    __bf16* vtb = (__bf16*)(ws + 56 * MB);    // 8MB
    float*  ap  = (float*) (ws + 64 * MB);    // 16MB  (reused as yf)
    float*  x1f = (float*) (ws + 80 * MB);    // 16MB
    __bf16* hb  = (__bf16*)(ws + 96 * MB);    // 32MB
    __bf16* ob  = vb;
    __bf16* x1b = xb;
    float*  yf  = ap;
    float*  outf = (float*)d_out;

    const dim3 tb(32, 8);

    // 1) convert x -> bf16
    {
        int n = M_ROWS * D_MODEL;
        cvt_bf16_kernel<<<n / 256, 256, 0, stream>>>(x, xb, n);
    }

    // 2) transpose+convert all weights to [N][K] bf16
    wtrans_kernel<<<dim3(D_MODEL / 32, D_MODEL / 32), tb, 0, stream>>>(wq, wqT, D_MODEL, D_MODEL);
    wtrans_kernel<<<dim3(D_MODEL / 32, D_MODEL / 32), tb, 0, stream>>>(wk, wkT, D_MODEL, D_MODEL);
    wtrans_kernel<<<dim3(D_MODEL / 32, D_MODEL / 32), tb, 0, stream>>>(wv, wvT, D_MODEL, D_MODEL);
    wtrans_kernel<<<dim3(D_MODEL / 32, D_MODEL / 32), tb, 0, stream>>>(wo, woT, D_MODEL, D_MODEL);
    wtrans_kernel<<<dim3(FFN_DIM / 32, D_MODEL / 32), tb, 0, stream>>>(w1, w1T, D_MODEL, FFN_DIM);
    wtrans_kernel<<<dim3(D_MODEL / 32, FFN_DIM / 32), tb, 0, stream>>>(w2, w2T, FFN_DIM, D_MODEL);

    // 3) Q/K/V projections (bf16 out)
    {
        dim3 grid(D_MODEL / 128, M_ROWS / 128);
        gemm_kernel<0, 1><<<grid, 256, 0, stream>>>(xb, wqT, bq, nullptr, qb, M_ROWS, D_MODEL, D_MODEL);
        gemm_kernel<0, 1><<<grid, 256, 0, stream>>>(xb, wkT, bk, nullptr, kb, M_ROWS, D_MODEL, D_MODEL);
        gemm_kernel<0, 1><<<grid, 256, 0, stream>>>(xb, wvT, bv, nullptr, vb, M_ROWS, D_MODEL, D_MODEL);
    }

    // 4) per-head V transpose -> vt[(b*H+h)*64+d][L]
    vtrans_kernel<<<dim3(SEQ_L / 32, 2, BATCH * N_HEAD), tb, 0, stream>>>(vb, vtb);

    // 5) flash attention -> ob (bf16, head-merged [B*L][D])
    attn_kernel<<<dim3(BATCH * N_HEAD, SEQ_L / 64), 128, 0, stream>>>(qb, kb, vtb, ob);

    // 6) output projection (f32 out)
    gemm_kernel<0, 0><<<dim3(D_MODEL / 128, M_ROWS / 128), 256, 0, stream>>>(
        ob, woT, bo, ap, nullptr, M_ROWS, D_MODEL, D_MODEL);

    // 7) residual + LN1 -> x1f (f32) and x1b (bf16)
    ln_kernel<<<M_ROWS, 256, 0, stream>>>(ap, x, g1, b1, x1f, x1b);

    // 8) FFN1 with exact GELU epilogue (bf16 out)
    gemm_kernel<1, 1><<<dim3(FFN_DIM / 128, M_ROWS / 128), 256, 0, stream>>>(
        x1b, w1T, bf1, nullptr, hb, M_ROWS, FFN_DIM, D_MODEL);

    // 9) FFN2 (f32 out)
    gemm_kernel<0, 0><<<dim3(D_MODEL / 128, M_ROWS / 128), 256, 0, stream>>>(
        hb, w2T, bf2, yf, nullptr, M_ROWS, D_MODEL, FFN_DIM);

    // 10) residual + LN2 -> d_out (f32)
    ln_kernel<<<M_ROWS, 256, 0, stream>>>(yf, x1f, g2, b2, outf, nullptr);
}
